// AMGSRN_89094801588891
// MI455X (gfx1250) — matrix operand order repaired and checked by reading, not compiled, not run on hardware
//
#include <hip/hip_runtime.h>

#define N_GRIDS 64
#define DIM 64
#define N_PTS (1 << 19)

typedef __bf16 bf16_t;
typedef __attribute__((ext_vector_type(16))) __bf16 v16bf;
typedef __attribute__((ext_vector_type(8)))  __bf16 v8bf;
typedef __attribute__((ext_vector_type(8)))  float  v8f;

union Frag { v16bf v; v8bf h[2]; };

__device__ __forceinline__ bf16_t f2bf(float f) {
    unsigned u = __builtin_bit_cast(unsigned, f);
    unsigned rr = (u + 0x7FFFu + ((u >> 16) & 1u)) >> 16;
    unsigned short s = (unsigned short)rr;
    return __builtin_bit_cast(bf16_t, s);
}

__device__ __forceinline__ unsigned pack2bf(float a, float b) {
    unsigned ua = __builtin_bit_cast(unsigned, a);
    unsigned ub = __builtin_bit_cast(unsigned, b);
    ua = (ua + 0x7FFFu + ((ua >> 16) & 1u)) >> 16;
    ub = (ub + 0x7FFFu + ((ub >> 16) & 1u)) >> 16;
    return (ua & 0xFFFFu) | (ub << 16);
}

__global__ __launch_bounds__(256) void amgsrn_fused(
    const float* __restrict__ x,  const float* __restrict__ r,
    const float* __restrict__ s,  const float* __restrict__ t,
    const float* __restrict__ feat,
    const float* __restrict__ W0, const float* __restrict__ W1,
    const float* __restrict__ W2, float* __restrict__ out)
{
    __shared__ __align__(16) float  s_xf[N_GRIDS * 12];   // scaled rotation + translation
    __shared__ __align__(16) bf16_t s_w0[64 * 128];       // W0 row-major [n][k]
    __shared__ __align__(16) bf16_t s_w1[64 * 64];        // W1 row-major [n][k]
    __shared__ __align__(16) float  s_w2[64];
    __shared__ __align__(16) bf16_t s_feat[8][16 * 128];  // per-wave A tile (reused for h)

    const int tid  = threadIdx.x;
    const int wave = tid >> 5;
    const int lane = tid & 31;

    // ---- stage per-grid transforms: M = diag(s) * R(q), t ----
    if (tid < N_GRIDS) {
        const int g = tid;
        float qw = r[g*4+0], qx = r[g*4+1], qy = r[g*4+2], qz = r[g*4+3];
        float inv = rsqrtf(qw*qw + qx*qx + qy*qy + qz*qz);
        qw *= inv; qx *= inv; qy *= inv; qz *= inv;
        float sx = s[g*3+0], sy = s[g*3+1], sz = s[g*3+2];
        float R00 = 1.f - 2.f*(qy*qy + qz*qz), R01 = 2.f*(qx*qy - qw*qz), R02 = 2.f*(qx*qz + qw*qy);
        float R10 = 2.f*(qx*qy + qw*qz), R11 = 1.f - 2.f*(qx*qx + qz*qz), R12 = 2.f*(qy*qz - qw*qx);
        float R20 = 2.f*(qx*qz - qw*qy), R21 = 2.f*(qy*qz + qw*qx), R22 = 1.f - 2.f*(qx*qx + qy*qy);
        s_xf[g*12+0] = sx*R00; s_xf[g*12+1] = sx*R01; s_xf[g*12+ 2] = sx*R02; s_xf[g*12+ 3] = t[g*3+0];
        s_xf[g*12+4] = sy*R10; s_xf[g*12+5] = sy*R11; s_xf[g*12+ 6] = sy*R12; s_xf[g*12+ 7] = t[g*3+1];
        s_xf[g*12+8] = sz*R20; s_xf[g*12+9] = sz*R21; s_xf[g*12+10] = sz*R22; s_xf[g*12+11] = t[g*3+2];
    }
    // ---- stage MLP weights as bf16 ----
    for (int i = tid; i < 64 * 128; i += 256) s_w0[i] = f2bf(W0[i]);
    for (int i = tid; i < 64 * 64;  i += 256) s_w1[i] = f2bf(W1[i]);
    if (tid < 64) s_w2[tid] = W2[tid];
    __syncthreads();

    const int tileBase = blockIdx.x * 128 + wave * 16;
    const int m  = lane & 15;   // point row within the 16-row tile
    const int g2 = lane >> 4;   // lane half (A/B fragment K-interleave group)
    const int p  = tileBase + m;

    const float px = x[p*3+0], py = x[p*3+1], pz = x[p*3+2];
    bf16_t* fbuf = s_feat[wave];

    // ---- encoding: trilinear sample all 64 grids for this lane's point ----
    for (int i = 0; i < 32; ++i) {
        const int g = 2*i + g2;
        const float4* T = (const float4*)&s_xf[g * 12];
        float4 r0 = T[0], r1 = T[1], r2 = T[2];
        float tx = fmaf(r0.x, px, fmaf(r0.y, py, fmaf(r0.z, pz, r0.w)));
        float ty = fmaf(r1.x, px, fmaf(r1.y, py, fmaf(r1.z, pz, r1.w)));
        float tz = fmaf(r2.x, px, fmaf(r2.y, py, fmaf(r2.z, pz, r2.w)));
        float gx = (tx + 1.f) * 31.5f;
        float gy = (ty + 1.f) * 31.5f;
        float gz = (tz + 1.f) * 31.5f;
        float fx0 = floorf(gx), fy0 = floorf(gy), fz0 = floorf(gz);
        float fx = gx - fx0, fy = gy - fy0, fz = gz - fz0;
        int ix = (int)fx0, iy = (int)fy0, iz = (int)fz0;
        // per-axis corner weights with validity folded in (matches reference mask)
        float wx0 = ((unsigned)ix       < 64u) ? (1.f - fx) : 0.f;
        float wx1 = ((unsigned)(ix + 1) < 64u) ? fx : 0.f;
        float wy0 = ((unsigned)iy       < 64u) ? (1.f - fy) : 0.f;
        float wy1 = ((unsigned)(iy + 1) < 64u) ? fy : 0.f;
        float wz0 = ((unsigned)iz       < 64u) ? (1.f - fz) : 0.f;
        float wz1 = ((unsigned)(iz + 1) < 64u) ? fz : 0.f;
        int cx0 = min(max(ix,     0), 63), cx1 = min(max(ix + 1, 0), 63);
        int cy0 = min(max(iy,     0), 63), cy1 = min(max(iy + 1, 0), 63);
        int cz0 = min(max(iz,     0), 63), cz1 = min(max(iz + 1, 0), 63);
        const float* f0 = feat + (size_t)g * (2 * DIM * DIM * DIM);
        const float* f1 = f0 + DIM * DIM * DIM;
        int ry0z0 = (cz0 * 64 + cy0) * 64;
        int ry1z0 = (cz0 * 64 + cy1) * 64;
        int ry0z1 = (cz1 * 64 + cy0) * 64;
        int ry1z1 = (cz1 * 64 + cy1) * 64;
        float w00 = wy0 * wz0, w10 = wy1 * wz0, w01 = wy0 * wz1, w11 = wy1 * wz1;
        float a0 = 0.f, a1 = 0.f, c;
        c = wx0 * w00; a0 = fmaf(c, f0[ry0z0 + cx0], a0); a1 = fmaf(c, f1[ry0z0 + cx0], a1);
        c = wx1 * w00; a0 = fmaf(c, f0[ry0z0 + cx1], a0); a1 = fmaf(c, f1[ry0z0 + cx1], a1);
        c = wx0 * w10; a0 = fmaf(c, f0[ry1z0 + cx0], a0); a1 = fmaf(c, f1[ry1z0 + cx0], a1);
        c = wx1 * w10; a0 = fmaf(c, f0[ry1z0 + cx1], a0); a1 = fmaf(c, f1[ry1z0 + cx1], a1);
        c = wx0 * w01; a0 = fmaf(c, f0[ry0z1 + cx0], a0); a1 = fmaf(c, f1[ry0z1 + cx0], a1);
        c = wx1 * w01; a0 = fmaf(c, f0[ry0z1 + cx1], a0); a1 = fmaf(c, f1[ry0z1 + cx1], a1);
        c = wx0 * w11; a0 = fmaf(c, f0[ry1z1 + cx0], a0); a1 = fmaf(c, f1[ry1z1 + cx0], a1);
        c = wx1 * w11; a0 = fmaf(c, f0[ry1z1 + cx1], a0); a1 = fmaf(c, f1[ry1z1 + cx1], a1);
        // feature order k = g*2 + c  -> pack two bf16, one 4B LDS store
        *(unsigned*)&fbuf[m * 128 + g * 2] = pack2bf(a0, a1);
    }

    // ---- layer 0: [16x128] @ [128x64] via 16x16x32 bf16 WMMA ----
    Frag A[4];
#pragma unroll
    for (int kc = 0; kc < 4; ++kc) {   // A frag: lo halves k=kc*32+8*g2, hi at +16
        const v8bf* bp = (const v8bf*)&fbuf[m * 128 + kc * 32 + g2 * 8];
        A[kc].h[0] = bp[0];
        A[kc].h[1] = bp[2];
    }
    bf16_t* hbuf = fbuf;  // reuse tile for hidden activations [16][64]
#pragma unroll
    for (int nt = 0; nt < 4; ++nt) {
        v8f acc = {};
        const int n = (lane & 15) + nt * 16;
#pragma unroll
        for (int kc = 0; kc < 4; ++kc) {
            Frag B;  // B frag: W0 row n, lo halves k=kc*32+8*g2, hi at +16
            const v8bf* wp = (const v8bf*)&s_w0[n * 128 + kc * 32 + g2 * 8];
            B.h[0] = wp[0]; B.h[1] = wp[2];
            acc = __builtin_amdgcn_wmma_f32_16x16x32_bf16(
                false, A[kc].v, false, B.v, (short)0, acc, false, false);
        }
#pragma unroll
        for (int rr = 0; rr < 8; ++rr) {  // ReLU + C-layout -> A-layout via LDS
            float v = acc[rr] > 0.f ? acc[rr] : 0.f;
            const int M = rr + g2 * 8;
            hbuf[M * 64 + n] = f2bf(v);
        }
    }

    // ---- layer 1: [16x64] @ [64x64], then layer 2 dot with W2 ----
    Frag A1[2];
#pragma unroll
    for (int kc = 0; kc < 2; ++kc) {
        const v8bf* bp = (const v8bf*)&hbuf[m * 64 + kc * 32 + g2 * 8];
        A1[kc].h[0] = bp[0];
        A1[kc].h[1] = bp[2];
    }
    float partial[8];
#pragma unroll
    for (int rr = 0; rr < 8; ++rr) partial[rr] = 0.f;
#pragma unroll
    for (int nt = 0; nt < 4; ++nt) {
        v8f acc = {};
        const int n = (lane & 15) + nt * 16;
#pragma unroll
        for (int kc = 0; kc < 2; ++kc) {
            Frag B;
            const v8bf* wp = (const v8bf*)&s_w1[n * 64 + kc * 32 + g2 * 8];
            B.h[0] = wp[0]; B.h[1] = wp[2];
            acc = __builtin_amdgcn_wmma_f32_16x16x32_bf16(
                false, A1[kc].v, false, B.v, (short)0, acc, false, false);
        }
        const float w2v = s_w2[n];
#pragma unroll
        for (int rr = 0; rr < 8; ++rr) {
            float hv = acc[rr] > 0.f ? acc[rr] : 0.f;
            partial[rr] = fmaf(hv, w2v, partial[rr]);
        }
    }
    // reduce the 16 N-columns held across the 16-lane half
#pragma unroll
    for (int rr = 0; rr < 8; ++rr) {
        float v = partial[rr];
        v += __shfl_xor(v, 1, 32);
        v += __shfl_xor(v, 2, 32);
        v += __shfl_xor(v, 4, 32);
        v += __shfl_xor(v, 8, 32);
        if ((lane & 15) == 0) out[tileBase + g2 * 8 + rr] = v;
    }
}

extern "C" void kernel_launch(void* const* d_in, const int* in_sizes, int n_in,
                              void* d_out, int out_size, void* d_ws, size_t ws_size,
                              hipStream_t stream) {
    const float* x    = (const float*)d_in[0];
    const float* r    = (const float*)d_in[1];
    const float* s    = (const float*)d_in[2];
    const float* t    = (const float*)d_in[3];
    const float* feat = (const float*)d_in[4];
    const float* W0   = (const float*)d_in[5];
    const float* W1   = (const float*)d_in[6];
    const float* W2   = (const float*)d_in[7];
    float* out = (float*)d_out;
    (void)in_sizes; (void)n_in; (void)out_size; (void)d_ws; (void)ws_size;
    amgsrn_fused<<<dim3(N_PTS / 128), dim3(256), 0, stream>>>(
        x, r, s, t, feat, W0, W1, W2, out);
}
